// ContextualHistoryEncoder_28836410425657
// MI455X (gfx1250) — matrix operand order repaired and checked by reading, not compiled, not run on hardware
//
#include <hip/hip_runtime.h>
#include <hip/hip_bf16.h>

// ---------------------------------------------------------------------------
// ContextualHistoryEncoder for MI455X (gfx1250, wave32, WMMA)
//
// Pipeline:
//   1. pack kernels      : f32 weights -> bf16 WMMA-B-tile layout (+gate perm)
//   2. encoder_kernel    : conv1+conv2 (VALU) + small encoders; writes conv2
//                          activations and feat tail in WMMA-A packed layout
//   3. gemm obs          : [T,1600]x[1600,128] WMMA -> featA (packed)
//   4. gemm pre0         : [T,192]x[192,1024] WMMA -> pre (f32, +bias)
//   5. recur layer0      : persistent-VGPR-weight WMMA recurrence (1 WGP),
//                          emits hs0 in A-packed layout; pre[] streamed with
//                          GLOBAL_LOAD_ASYNC_TO_LDS (ASYNCcnt) double buffer
//   6. gemm pre1         : [T,256]x[256,1024] WMMA -> pre
//   7. recur layer1      : same, writes embedding/h1/c1
// ---------------------------------------------------------------------------

typedef __bf16 bf16;
typedef bf16  v16bf __attribute__((ext_vector_type(16)));
typedef float v8f   __attribute__((ext_vector_type(8)));

#define T_LEN 16384

// ---- CDNA5 async-to-LDS path (guarded: falls back to VGPR staging) --------
#if defined(__has_builtin)
#  if __has_builtin(__builtin_amdgcn_global_load_async_to_lds_b32)
#    define HAVE_ASYNC_LDS 1
#  endif
#  if __has_builtin(__builtin_amdgcn_s_wait_asynccnt)
#    define HAVE_WAIT_ASYNC 1
#  endif
#endif

#if defined(HAVE_ASYNC_LDS)
// param 0: global (address_space(1)) int*  [source]
// param 1: LDS    (address_space(3)) int*  [destination]
#define ASYNC_CP_B32(dst_lds, src_glb)                                  \
  __builtin_amdgcn_global_load_async_to_lds_b32(                        \
      (__attribute__((address_space(1))) int*)(src_glb),                \
      (__attribute__((address_space(3))) int*)(dst_lds), 0, 0)
#endif

__device__ __forceinline__ void async_wait_all() {
#if defined(HAVE_WAIT_ASYNC)
  __builtin_amdgcn_s_wait_asynccnt(0);
#elif defined(HAVE_ASYNC_LDS)
  asm volatile("s_wait_asynccnt 0x0" ::: "memory");
#endif
}

// ---- WMMA 16x16x32 bf16 operand lane layouts (per CDNA5 ISA 7.12.2) -------
// A tile (16Mx32K): lane, half -> (M = lane%16, K below). Inverse map:
__device__ __forceinline__ int a_slot(int m, int ko) {
  int hh, l16;
  if (ko < 8)       { hh = ko;      l16 = 0; }
  else if (ko < 16) { hh = ko - 8;  l16 = 1; }
  else if (ko < 24) { hh = ko - 8;  l16 = 0; }
  else              { hh = ko - 16; l16 = 1; }
  return (m + 16 * l16) * 16 + hh;       // element index within 512-elem tile
}

// Gate-row permutation: wave w owns hidden slice j in [8w,8w+8) for all 4
// gates; n' = 32w + gate*8 + s  ->  original row = gate*256 + 8w + s.
__device__ __forceinline__ int perm_row(int n) {
  int w = n >> 5, r = n & 31, gate = r >> 3, s8 = r & 7;
  return gate * 256 + w * 8 + s8;
}

__device__ __forceinline__ float sigm(float x) { return 1.f / (1.f + __expf(-x)); }

// ---------------------------------------------------------------------------
// Pack W[n][k] (f32) into bf16 B-operand tiles: dst[(kt*NTC+nt)*512 + l*16+h]
// B tile (32Kx16N): lane l, half h -> N = l%16, K = (h&8)*2 + (l/16)*8 + (h&7).
// ---------------------------------------------------------------------------
__global__ __launch_bounds__(512) void pack_b_kernel(
    const float* __restrict__ W, int Ksrc, int NTC, int doPerm,
    bf16* __restrict__ dst) {
  const int tile = blockIdx.x;           // kt*NTC + nt
  const int kt = tile / NTC, nt = tile % NTC;
  const int l = threadIdx.x >> 4, hh = threadIdx.x & 15;
  const int n = nt * 16 + (l & 15);
  const int k = kt * 32 + ((hh & 8) << 1) + ((l >> 4) << 3) + (hh & 7);
  const int row = doPerm ? perm_row(n) : n;
  float v = (k < Ksrc) ? W[(size_t)row * Ksrc + k] : 0.f;
  dst[(size_t)tile * 512 + threadIdx.x] = (bf16)v;
}

__global__ __launch_bounds__(256) void pack_bias_kernel(
    const float* __restrict__ b1, const float* __restrict__ b2,
    float* __restrict__ dst) {
  int n = blockIdx.x * 256 + threadIdx.x;       // 1024
  int row = perm_row(n);
  dst[n] = b1[row] + b2[row];
}

// ---------------------------------------------------------------------------
// Encoder: 4 waves/block, one timestep per wave. conv1/conv2 in VALU (tiny),
// conv2 output + feature tail written directly in WMMA-A packed bf16 layout.
// ---------------------------------------------------------------------------
__device__ __forceinline__ void featW(bf16* featA, int mt, int m, int k, float v) {
  featA[((size_t)mt * 6 + (k >> 5)) * 512 + a_slot(m, k & 31)] = (bf16)v;
}

__global__ __launch_bounds__(128) void encoder_kernel(
    const float* __restrict__ fov, const float* __restrict__ own_pos,
    const float* __restrict__ oppo, const float* __restrict__ cost,
    const int* __restrict__ action,
    const float* __restrict__ c1w, const float* __restrict__ c1b,
    const float* __restrict__ c2w, const float* __restrict__ c2b,
    const float* __restrict__ pos_w, const float* __restrict__ pos_b,
    const float* __restrict__ opp_w, const float* __restrict__ opp_b,
    const float* __restrict__ act_emb,
    bf16* __restrict__ c2A, bf16* __restrict__ featA) {
  __shared__ float sFov[4][80];
  __shared__ float sC1[4][25 * 32];
  __shared__ float sC2[4][25 * 64];
  const int lane = threadIdx.x & 31;
  const int wave = threadIdx.x >> 5;
  const int t = blockIdx.x * 4 + wave;
  const int mt = t >> 4, m = t & 15;

  for (int i = lane; i < 75; i += 32) sFov[wave][i] = fov[(size_t)t * 75 + i];
  __syncthreads();

  // conv1: lane = output channel (32)
  float w1[27];
#pragma unroll
  for (int i = 0; i < 27; ++i) w1[i] = c1w[lane * 27 + i];
  const float b1 = c1b[lane];
  for (int pos = 0; pos < 25; ++pos) {
    const int y = pos / 5, x = pos % 5;
    float acc = b1;
#pragma unroll
    for (int ky = 0; ky < 3; ++ky) {
      const int iy = y + ky - 1;
      if (iy < 0 || iy > 4) continue;
#pragma unroll
      for (int kx = 0; kx < 3; ++kx) {
        const int ix = x + kx - 1;
        if (ix < 0 || ix > 4) continue;
#pragma unroll
        for (int ci = 0; ci < 3; ++ci)
          acc += sFov[wave][(iy * 5 + ix) * 3 + ci] * w1[ci * 9 + ky * 3 + kx];
      }
    }
    sC1[wave][pos * 32 + lane] = fmaxf(acc, 0.f);
  }
  __syncthreads();

  // conv2: lane handles channels lane and lane+32 (weights L1/L2 resident)
  for (int o2 = 0; o2 < 2; ++o2) {
    const int oc = lane + 32 * o2;
    const float* wgt = c2w + (size_t)oc * 32 * 9;
    const float bb = c2b[oc];
    for (int pos = 0; pos < 25; ++pos) {
      const int y = pos / 5, x = pos % 5;
      float acc = bb;
      for (int ky = 0; ky < 3; ++ky) {
        const int iy = y + ky - 1;
        if (iy < 0 || iy > 4) continue;
        for (int kx = 0; kx < 3; ++kx) {
          const int ix = x + kx - 1;
          if (ix < 0 || ix > 4) continue;
          const float* c1p = &sC1[wave][(iy * 5 + ix) * 32];
#pragma unroll
          for (int ci = 0; ci < 32; ++ci)
            acc += c1p[ci] * wgt[ci * 9 + ky * 3 + kx];
        }
      }
      sC2[wave][pos * 64 + oc] = fmaxf(acc, 0.f);
    }
  }
  __syncthreads();

  // conv2 activations -> WMMA-A packed bf16 (flatten K = ch*25 + pos, K=1600)
  for (int o2 = 0; o2 < 2; ++o2) {
    const int oc = lane + 32 * o2;
    for (int pos = 0; pos < 25; ++pos) {
      const int k = oc * 25 + pos;
      c2A[((size_t)mt * 50 + (k >> 5)) * 512 + a_slot(m, k & 31)] =
          (bf16)sC2[wave][pos * 64 + oc];
    }
  }

  // small encoders + padding -> featA columns 128..191 (cols 0..127 = obs GEMM)
  if (lane < 16) {
    const float px = own_pos[t * 2 + 0] * (1.f / 25.f);
    const float py = own_pos[t * 2 + 1] * (1.f / 25.f);
    float v = pos_b[lane] + px * pos_w[lane * 2 + 0] + py * pos_w[lane * 2 + 1];
    featW(featA, mt, m, 128 + lane, v);

    const float q0 = oppo[t * 3 + 0] * (1.f / 25.f);
    const float q1 = oppo[t * 3 + 1] * (1.f / 25.f);
    const float q2 = oppo[t * 3 + 2] * (1.f / 150.f);
    v = opp_b[lane] + q0 * opp_w[lane * 3 + 0] + q1 * opp_w[lane * 3 + 1] +
        q2 * opp_w[lane * 3 + 2];
    featW(featA, mt, m, 144 + lane, v);

    featW(featA, mt, m, 160 + lane, act_emb[action[t] * 16 + lane]);
    featW(featA, mt, m, 176 + lane, (lane == 0) ? cost[t] : 0.f);  // + zero pad
  }
}

// ---------------------------------------------------------------------------
// Generic bf16 WMMA GEMM on pre-packed operands. Ping-pong software pipeline
// (unroll-by-2, two operand register sets): loads for one set issue while the
// WMMAs of the other set execute. KT/NTC are compile-time so all pipeline
// guards constant-fold into a straight-line loop with partial loadcnt waits.
//   Ap: [mt][KT][512] bf16 (A-layout), Bp: [kt][NTC][512] bf16 (B-layout)
//   NTPW = N-tiles per wave (8 waves/block). Output: f32 [16*grid][ldn] or
//   bf16 A-packed (PACK_OUT) for feeding the next GEMM. KT must be even.
// ---------------------------------------------------------------------------
template <int NTPW, int KT, int NTC, bool PACK_OUT>
__global__ __launch_bounds__(256) void gemm_kernel(
    const bf16* __restrict__ Ap, const bf16* __restrict__ Bp,
    const float* __restrict__ bias,
    float* __restrict__ outF, int ldn, bf16* __restrict__ outP, int dstKT) {
  const int lane = threadIdx.x & 31;
  const int wave = threadIdx.x >> 5;
  const int mt = blockIdx.x;

  const bf16* __restrict__ aBase = Ap + (size_t)mt * KT * 512 + lane * 16;
  const bf16* __restrict__ bBase =
      Bp + (size_t)(wave * NTPW) * 512 + lane * 16;

  v8f acc[NTPW];
#pragma unroll
  for (int i = 0; i < NTPW; ++i) {
    const float b = bias[(wave * NTPW + i) * 16 + (lane & 15)];
#pragma unroll
    for (int r = 0; r < 8; ++r) acc[i][r] = b;
  }

  // slot 0 <- kt = 0
  v16bf a0 = *(const v16bf*)(aBase);
  v16bf b0[NTPW];
#pragma unroll
  for (int i = 0; i < NTPW; ++i)
    b0[i] = *(const v16bf*)(bBase + (size_t)i * 512);
  v16bf a1 = a0;
  v16bf b1[NTPW];
#pragma unroll
  for (int i = 0; i < NTPW; ++i) b1[i] = b0[i];

  for (int kt = 0; kt < KT; kt += 2) {
    // slot 1 <- kt+1 (in flight during slot-0 WMMAs); KT even => always valid
    a1 = *(const v16bf*)(aBase + (size_t)(kt + 1) * 512);
#pragma unroll
    for (int i = 0; i < NTPW; ++i)
      b1[i] = *(const v16bf*)(bBase + ((size_t)(kt + 1) * NTC + i) * 512);
    if (kt + 3 < KT) {  // prefetch the K-slice after the next load group
      __builtin_prefetch(aBase + (size_t)(kt + 3) * 512, 0, 3);
      __builtin_prefetch(bBase + (size_t)(kt + 3) * NTC * 512, 0, 3);
    }
#pragma unroll
    for (int i = 0; i < NTPW; ++i)
      acc[i] = __builtin_amdgcn_wmma_f32_16x16x32_bf16(
          false, a0, false, b0[i], (short)0, acc[i], false, false);
    if (kt + 2 < KT) {  // slot 0 <- kt+2 (in flight during slot-1 WMMAs)
      a0 = *(const v16bf*)(aBase + (size_t)(kt + 2) * 512);
#pragma unroll
      for (int i = 0; i < NTPW; ++i)
        b0[i] = *(const v16bf*)(bBase + ((size_t)(kt + 2) * NTC + i) * 512);
    }
#pragma unroll
    for (int i = 0; i < NTPW; ++i)
      acc[i] = __builtin_amdgcn_wmma_f32_16x16x32_bf16(
          false, a1, false, b1[i], (short)0, acc[i], false, false);
  }

#pragma unroll
  for (int i = 0; i < NTPW; ++i) {
    const int nt = wave * NTPW + i;
#pragma unroll
    for (int r = 0; r < 8; ++r) {
      const int mm = r + 8 * (lane >> 4);
      const int n = nt * 16 + (lane & 15);
      const float v = acc[i][r];
      if (PACK_OUT) {
        outP[((size_t)mt * dstKT + (n >> 5)) * 512 + a_slot(mm, n & 31)] =
            (bf16)v;
      } else {
        outF[(size_t)(mt * 16 + mm) * ldn + n] = v;
      }
    }
  }
}

// ---------------------------------------------------------------------------
// LSTM recurrence: 1 block, 32 waves. whh (gate-permuted, bf16) lives in
// VGPRs for the whole sequence: wave w owns 2 N-tiles (4 gates x 8 hidden),
// 2x8 v16bf = 128 VGPRs/lane. Per step: 16 WMMAs/wave, then the 256 combine
// threads write the new h directly into the M-replicated A-operand LDS tile
// (each thread owns one K index -> 16 slots), giving 2 barriers per step.
// pre[] streamed into an LDS double buffer via GLOBAL_LOAD_ASYNC_TO_LDS
// (ASYNCcnt), one 4-step chunk ahead.
// ---------------------------------------------------------------------------
template <bool STORE_HS>
__global__ __launch_bounds__(1024, 1) void recur_kernel(
    const float* __restrict__ pre,   // [T][1024] gate-permuted, bias included
    const bf16* __restrict__ Wrp,    // [8][64][512] packed B tiles
    bf16* __restrict__ hsA,          // [T/16][8][512] A-packed hs (layer 0)
    float* __restrict__ outH, float* __restrict__ outC,
    float* __restrict__ outEmb) {
  __shared__ __align__(32) bf16 hA[8 * 512];     // h as A operand, M-replicated
  __shared__ float gbuf[1024];
  __shared__ float preChunk[2][4 * 1024];

  const int tid = threadIdx.x;
  const int lane = tid & 31;
  const int wave = tid >> 5;

  // persistent recurrent weights in VGPRs
  v16bf wB[2][8];
#pragma unroll
  for (int j = 0; j < 2; ++j)
#pragma unroll
    for (int kt = 0; kt < 8; ++kt)
      wB[j][kt] = *(const v16bf*)(Wrp +
                  ((size_t)(kt * 64 + (wave * 2 + j)) * 512 + lane * 16));

  for (int i = tid; i < 8 * 512; i += 1024) hA[i] = (bf16)0.0f;
  float cReg = 0.f, hReg = 0.f;

  // per-combine-thread hA slot base (thread j owns K index j)
  const int jkt = (tid & 255) >> 5, jko = tid & 31;
  int jhh, jl16;
  if (jko < 8)       { jhh = jko;      jl16 = 0; }
  else if (jko < 16) { jhh = jko - 8;  jl16 = 1; }
  else if (jko < 24) { jhh = jko - 8;  jl16 = 0; }
  else               { jhh = jko - 16; jl16 = 1; }
  bf16* const hDst = hA + jkt * 512 + jl16 * 256 + jhh;

  // stage chunk 0 of pre[] into LDS
#if defined(HAVE_ASYNC_LDS)
#pragma unroll
  for (int s = 0; s < 4; ++s)
    ASYNC_CP_B32(&preChunk[0][s * 1024 + tid], pre + (size_t)s * 1024 + tid);
  async_wait_all();
#else
  float pr[4];
#pragma unroll
  for (int s = 0; s < 4; ++s) pr[s] = pre[(size_t)s * 1024 + tid];
#pragma unroll
  for (int s = 0; s < 4; ++s) preChunk[0][s * 1024 + tid] = pr[s];
#endif
  __syncthreads();

  const int NCH = T_LEN / 4;
  for (int ch = 0; ch < NCH; ++ch) {
    const int cur = ch & 1;
    if (ch + 1 < NCH) {  // stream next chunk behind this chunk's compute
#if defined(HAVE_ASYNC_LDS)
#pragma unroll
      for (int s = 0; s < 4; ++s)
        ASYNC_CP_B32(&preChunk[cur ^ 1][s * 1024 + tid],
                     pre + ((size_t)(ch + 1) * 4 + s) * 1024 + tid);
#else
#pragma unroll
      for (int s = 0; s < 4; ++s)
        pr[s] = pre[((size_t)(ch + 1) * 4 + s) * 1024 + tid];
#endif
    }
#pragma unroll 1
    for (int s = 0; s < 4; ++s) {
      const int t = ch * 4 + s;
      v8f acc0, acc1;
#pragma unroll
      for (int r = 0; r < 8; ++r) { acc0[r] = 0.f; acc1[r] = 0.f; }
#pragma unroll
      for (int kt = 0; kt < 8; ++kt) {
        const v16bf aop = *(const v16bf*)(hA + kt * 512 + lane * 16);
        acc0 = __builtin_amdgcn_wmma_f32_16x16x32_bf16(
            false, aop, false, wB[0][kt], (short)0, acc0, false, false);
        acc1 = __builtin_amdgcn_wmma_f32_16x16x32_bf16(
            false, aop, false, wB[1][kt], (short)0, acc1, false, false);
      }
      if (lane < 16) {  // z = matvec + pre  (all C rows identical; row 0)
        const int n0 = wave * 32 + lane;
        gbuf[n0]      = acc0[0] + preChunk[cur][s * 1024 + n0];
        gbuf[n0 + 16] = acc1[0] + preChunk[cur][s * 1024 + n0 + 16];
      }
      __syncthreads();
      if (tid < 256) {  // gate combine; thread j owns hidden j = K index j
        const int j = tid, w = j >> 3, s8 = j & 7;
        const float zi = gbuf[w * 32 + s8];
        const float zf = gbuf[w * 32 + 8 + s8];
        const float zg = gbuf[w * 32 + 16 + s8];
        const float zo = gbuf[w * 32 + 24 + s8];
        cReg = sigm(zf) * cReg + sigm(zi) * tanhf(zg);
        hReg = sigm(zo) * tanhf(cReg);
        const bf16 hb = (bf16)hReg;
        // write h[j] into all 16 replicated M-rows of its A-tile slot
#pragma unroll
        for (int mi = 0; mi < 16; ++mi) hDst[mi * 16] = hb;
        if (STORE_HS) {
          const int mm = t & 15, mt = t >> 4;
          hsA[((size_t)(mt * 8 + (j >> 5))) * 512 + a_slot(mm, j & 31)] = hb;
        }
      }
      __syncthreads();
    }
#if !defined(HAVE_ASYNC_LDS)
    if (ch + 1 < NCH) {
#pragma unroll
      for (int s = 0; s < 4; ++s) preChunk[cur ^ 1][s * 1024 + tid] = pr[s];
    }
#endif
    async_wait_all();   // next chunk's LDS data resident before flip
    __syncthreads();
  }

  if (tid < 256) {
    outH[tid] = hReg;
    outC[tid] = cReg;
    if (outEmb) outEmb[tid] = hReg;
  }
}

// ---------------------------------------------------------------------------
extern "C" void kernel_launch(void* const* d_in, const int* in_sizes, int n_in,
                              void* d_out, int out_size, void* d_ws,
                              size_t ws_size, hipStream_t stream) {
  const float* fov     = (const float*)d_in[0];
  const float* own_pos = (const float*)d_in[1];
  const float* oppo    = (const float*)d_in[2];
  const float* cost    = (const float*)d_in[3];
  const int*   action  = (const int*)d_in[4];
  const float* c1w = (const float*)d_in[5];
  const float* c1b = (const float*)d_in[6];
  const float* c2w = (const float*)d_in[7];
  const float* c2b = (const float*)d_in[8];
  const float* obs_w = (const float*)d_in[9];
  const float* obs_b = (const float*)d_in[10];
  const float* pos_w = (const float*)d_in[11];
  const float* pos_b = (const float*)d_in[12];
  const float* opp_w = (const float*)d_in[13];
  const float* opp_b = (const float*)d_in[14];
  const float* act_emb = (const float*)d_in[15];
  const float* wih0 = (const float*)d_in[16];
  const float* whh0 = (const float*)d_in[17];
  const float* bih0 = (const float*)d_in[18];
  const float* bhh0 = (const float*)d_in[19];
  const float* wih1 = (const float*)d_in[20];
  const float* whh1 = (const float*)d_in[21];
  const float* bih1 = (const float*)d_in[22];
  const float* bhh1 = (const float*)d_in[23];
  float* out = (float*)d_out;

  // workspace layout (all sizes 256B-multiples)
  char* ws = (char*)d_ws;
  size_t off = 0;
  bf16* featA = (bf16*)(ws + off); off += (size_t)1024 * 6 * 512 * 2;   // 6.3MB
  bf16* c2A   = (bf16*)(ws + off); off += (size_t)1024 * 50 * 512 * 2;  // 52MB
  bf16* hs0A  = (bf16*)(ws + off); off += (size_t)1024 * 8 * 512 * 2;   // 8.4MB
  bf16* Wb0p  = (bf16*)(ws + off); off += (size_t)6 * 64 * 512 * 2;
  bf16* Wb1p  = (bf16*)(ws + off); off += (size_t)8 * 64 * 512 * 2;
  bf16* Wr0p  = (bf16*)(ws + off); off += (size_t)8 * 64 * 512 * 2;
  bf16* Wr1p  = (bf16*)(ws + off); off += (size_t)8 * 64 * 512 * 2;
  bf16* ObsBp = (bf16*)(ws + off); off += (size_t)50 * 8 * 512 * 2;
  float* bsum0 = (float*)(ws + off); off += 4096;
  float* bsum1 = (float*)(ws + off); off += 4096;
  float* pre   = (float*)(ws + off); off += (size_t)T_LEN * 1024 * 4;   // 64MB (shared pre0/pre1)

  // 1) weight packing
  pack_b_kernel<<<50 * 8, 512, 0, stream>>>(obs_w, 1600, 8, 0, ObsBp);
  pack_b_kernel<<<6 * 64, 512, 0, stream>>>(wih0, 177, 64, 1, Wb0p);
  pack_b_kernel<<<8 * 64, 512, 0, stream>>>(wih1, 256, 64, 1, Wb1p);
  pack_b_kernel<<<8 * 64, 512, 0, stream>>>(whh0, 256, 64, 1, Wr0p);
  pack_b_kernel<<<8 * 64, 512, 0, stream>>>(whh1, 256, 64, 1, Wr1p);
  pack_bias_kernel<<<4, 256, 0, stream>>>(bih0, bhh0, bsum0);
  pack_bias_kernel<<<4, 256, 0, stream>>>(bih1, bhh1, bsum1);

  // 2) per-step encoders (batch over T)
  encoder_kernel<<<T_LEN / 4, 128, 0, stream>>>(
      fov, own_pos, oppo, cost, action, c1w, c1b, c2w, c2b, pos_w, pos_b,
      opp_w, opp_b, act_emb, c2A, featA);

  // 3) obs linear: [T,1600]x[1600,128] -> featA cols 0..127 (packed bf16)
  gemm_kernel<1, 50, 8, true><<<T_LEN / 16, 256, 0, stream>>>(
      c2A, ObsBp, obs_b, nullptr, 0, featA, 6);

  // 4) layer-0 input projection: [T,192]x[192,1024] -> pre
  gemm_kernel<8, 6, 64, false><<<T_LEN / 16, 256, 0, stream>>>(
      featA, Wb0p, bsum0, pre, 1024, nullptr, 0);

  // 5) layer-0 recurrence (writes h0 -> out[256:512], c0 -> out[768:1024])
  recur_kernel<true><<<1, 1024, 0, stream>>>(pre, Wr0p, hs0A, out + 256,
                                             out + 768, (float*)nullptr);

  // 6) layer-1 input projection: [T,256]x[256,1024] -> pre
  gemm_kernel<8, 8, 64, false><<<T_LEN / 16, 256, 0, stream>>>(
      hs0A, Wb1p, bsum1, pre, 1024, nullptr, 0);

  // 7) layer-1 recurrence (embedding -> out[0:256], h1 -> out[512:768],
  //    c1 -> out[1024:1280])
  recur_kernel<false><<<1, 1024, 0, stream>>>(pre, Wr1p, (bf16*)nullptr,
                                              out + 512, out + 1024, out);
}
